// LSTM_1090921693292
// MI455X (gfx1250) — matrix operand order repaired and checked by reading, not compiled, not run on hardware
//
#include <hip/hip_runtime.h>
#include <hip/hip_bf16.h>

// ---------------- problem constants ----------------
constexpr int kT    = 400;
constexpr int kB    = 8;
constexpr int kB2   = 16;      // bidirectional batch (fwd + reversed)
constexpr int kD0   = 440;     // input dim
constexpr int kD0P  = 448;     // padded to multiple of 32 for WMMA K
constexpr int kH    = 1024;
constexpr int kG4   = 4 * kH;  // 4 gates concatenated: f,i,o,c
constexpr int kD1   = 2 * kH;  // layer >=1 input dim
constexpr int kC    = 3000;    // classes
constexpr int kRows = kT * kB; // classifier rows
constexpr float kDrop = 0.8f;  // 1 - DROP_RATE (test path)
constexpr float kEps  = 1e-6f;

typedef __attribute__((ext_vector_type(16))) __bf16 bf16x16;
typedef __attribute__((ext_vector_type(8)))  float  f32x8;
typedef __hip_bfloat16 bf16_t;

// ---------------- elementwise helpers ----------------
__device__ __forceinline__ float sigmoidf_(float x) { return 1.0f / (1.0f + __expf(-x)); }

// fp32 -> bf16 plain convert (grid-stride)
__global__ void k_f32_to_bf16(const float* __restrict__ s, bf16_t* __restrict__ d, long n) {
    long i = (long)blockIdx.x * blockDim.x + threadIdx.x;
    long stride = (long)gridDim.x * blockDim.x;
    for (; i < n; i += stride) d[i] = __float2bfloat16(s[i]);
}

// weight [rows, K] fp32 -> [rows, ldp] bf16 with zero K-padding
__global__ void k_w_pad_bf16(const float* __restrict__ w, bf16_t* __restrict__ d,
                             int rows, int K, int ldp) {
    long n = (long)rows * ldp;
    long i = (long)blockIdx.x * blockDim.x + threadIdx.x;
    long stride = (long)gridDim.x * blockDim.x;
    for (; i < n; i += stride) {
        int r = (int)(i / ldp), k = (int)(i % ldp);
        d[i] = (k < K) ? __float2bfloat16(w[(long)r * K + k]) : __float2bfloat16(0.0f);
    }
}

// concatenate the four gate biases (f,i,o,c) into one 4096 vector
__global__ void k_bias_concat(const float* __restrict__ bf_, const float* __restrict__ bi_,
                              const float* __restrict__ bo_, const float* __restrict__ bc_,
                              float* __restrict__ bcat) {
    int i = blockIdx.x * blockDim.x + threadIdx.x;
    if (i < kG4) {
        const float* src = (i < kH) ? bf_ : (i < 2 * kH) ? bi_ : (i < 3 * kH) ? bo_ : bc_;
        bcat[i] = src[i & (kH - 1)];
    }
}

// build bidirectional, bf16, K-padded input: xb[t][bb][ldp]
//   bb <  8 : x[t][bb]        (forward)
//   bb >= 8 : x[T-1-t][bb-8]  (time-reversed copy)
__global__ void k_build_xb(const float* __restrict__ x, bf16_t* __restrict__ xb,
                           int Din, int ldp) {
    long n = (long)kT * kB2 * ldp;
    long i = (long)blockIdx.x * blockDim.x + threadIdx.x;
    long stride = (long)gridDim.x * blockDim.x;
    for (; i < n; i += stride) {
        int t  = (int)(i / ((long)kB2 * ldp));
        int r  = (int)(i % ((long)kB2 * ldp));
        int bb = r / ldp, k = r % ldp;
        float v = 0.0f;
        if (k < Din) {
            int ts = (bb < kB) ? t : (kT - 1 - t);
            int bs = (bb < kB) ? bb : (bb - kB);
            v = x[((long)ts * kB + bs) * Din + k];
        }
        xb[i] = __float2bfloat16(v);
    }
}

// merge bidir hidden states into next-layer input: xn[t][b][2H]
//   k <  H : hs[t][b][k]            (forward half)
//   k >= H : hs[T-1-t][b+8][k-H]    (backward half, time-flipped back)
__global__ void k_merge_bidir(const float* __restrict__ hs, float* __restrict__ xn) {
    long n = (long)kT * kB * kD1;
    long i = (long)blockIdx.x * blockDim.x + threadIdx.x;
    long stride = (long)gridDim.x * blockDim.x;
    for (; i < n; i += stride) {
        int t = (int)(i / ((long)kB * kD1));
        int r = (int)(i % ((long)kB * kD1));
        int b = r / kD1, k = r % kD1;
        float v;
        if (k < kH) v = hs[((long)t * kB2 + b) * kH + k];
        else        v = hs[((long)(kT - 1 - t) * kB2 + (kB + b)) * kH + (k - kH)];
        xn[i] = v;
    }
}

// ---------------- WMMA bf16 GEMM:  C[m][n] = sum_k A[m][k]*W[n][k] + bias[n] ----------------
// One wave computes a 16 x (16*NT) tile: the A fragment is loaded once per k-step and
// reused across NT v_wmma_f32_16x16x32_bf16 issues (raises FLOP/byte from ~8 to ~12.8 at NT=4).
// The k-loop is explicitly software-pipelined (double-buffered fragments): loads for
// iteration k+1 are issued BEFORE the WMMAs consuming iteration k, so the next-iteration
// loads are live across the WMMA block, forcing a second register set and partial
// s_wait_loadcnt (loads stay in flight under the WMMAs instead of wait 0x0 per WMMA).
// Per-lane operand layout (CDNA5 ISA 7.12.2, 16-bit A 16x32 / B 32x16):
//   A: lane L (g=L>>4, r=L&15) row m0+r, K chunks [k0+8g .. +7] and [k0+16+8g .. +7]
//   B: lane L col n0+j*16+r (W row), K chunks [k0+8g .. +7] and [k0+16+8g .. +7]
// All fragments are 16B-aligned contiguous runs -> global_load_b128 pairs.
template <int NT>
__global__ void k_gemm_bf16(const bf16_t* __restrict__ A, int lda,
                            const bf16_t* __restrict__ W, int ldw,
                            const float* __restrict__ bias,
                            float* __restrict__ C, int ldc,
                            int M, int N, int K) {
    const int lane = threadIdx.x & 31;
    const int wave = threadIdx.x >> 5;
    const int tilesC = (N + 16 * NT - 1) / (16 * NT);
    const int tilesM = M >> 4;                 // M is always a multiple of 16 here
    const int tile = blockIdx.x * (blockDim.x >> 5) + wave;
    if (tile >= tilesM * tilesC) return;       // wave-uniform exit (EXEC stays full for WMMA)
    const int tm = tile / tilesC, tc = tile % tilesC;
    const int m0 = tm << 4, n0 = tc * (16 * NT);
    const int g = lane >> 4, r = lane & 15;

    const bf16_t* Ap = A + (long)(m0 + r) * lda + 8 * g;       // lane A base (chunk 0)
    const bf16_t* Wp[NT];
#pragma unroll
    for (int j = 0; j < NT; ++j) {
        int brow = n0 + j * 16 + r;
        if (brow >= N) brow = N - 1;           // clamp pad rows (stores are guarded)
        Wp[j] = W + (long)brow * ldw + 8 * g;  // lane B base
    }

    f32x8 acc[NT];
#pragma unroll
    for (int j = 0; j < NT; ++j) acc[j] = (f32x8){};

    union Frag { bf16x16 v; uint4 q[2]; };

    // prologue: load k-block 0
    Frag a, b[NT];
    a.q[0] = *(const uint4*)(Ap + 0);
    a.q[1] = *(const uint4*)(Ap + 16);
#pragma unroll
    for (int j = 0; j < NT; ++j) {
        b[j].q[0] = *(const uint4*)(Wp[j] + 0);
        b[j].q[1] = *(const uint4*)(Wp[j] + 16);
    }

#pragma unroll 2
    for (int k0 = 0; k0 < K; k0 += 32) {
        const int kn = (k0 + 32 < K) ? (k0 + 32) : k0;   // last iter: harmless re-load
        // ---- issue next-iteration loads first (second register set, stays in flight) ----
        Frag an, bn[NT];
        an.q[0] = *(const uint4*)(Ap + kn);
        an.q[1] = *(const uint4*)(Ap + kn + 16);
#pragma unroll
        for (int j = 0; j < NT; ++j) {
            bn[j].q[0] = *(const uint4*)(Wp[j] + kn);
            bn[j].q[1] = *(const uint4*)(Wp[j] + kn + 16);
        }
        __builtin_prefetch(Ap + kn + 256, 0, 1);   // global_prefetch_b8, 256 elements (8 k-blocks) past the next load
        // ---- WMMA phase on current fragments ----
#pragma unroll
        for (int j = 0; j < NT; ++j) {
            acc[j] = __builtin_amdgcn_wmma_f32_16x16x32_bf16(
                /*neg_a=*/false, a.v, /*neg_b=*/false, b[j].v,
                /*c_mod=*/(short)0, acc[j], /*reuse_a=*/false, /*reuse_b=*/false);
        }
        // ---- rotate (renamed away by unroll-by-2) ----
        a = an;
#pragma unroll
        for (int j = 0; j < NT; ++j) b[j] = bn[j];
    }

#pragma unroll
    for (int j = 0; j < NT; ++j) {
        const int col = n0 + j * 16 + r;
        if (col < N) {
            const float bv = bias ? bias[col] : 0.0f;
#pragma unroll
            for (int v = 0; v < 8; ++v) {      // D row = 8*g + v (ISA C/D layout)
                C[(long)(m0 + 8 * g + v) * ldc + col] = acc[j][v] + bv;
            }
        }
    }
}

template <int NT>
static void launch_gemm(const bf16_t* A, int lda, const bf16_t* W, int ldw, const float* bias,
                        float* C, int ldc, int M, int N, int K, hipStream_t stream) {
    int tiles = (M >> 4) * ((N + 16 * NT - 1) / (16 * NT));
    int blocks = (tiles + 7) / 8;
    k_gemm_bf16<NT><<<blocks, 256, 0, stream>>>(A, lda, W, ldw, bias, C, ldc, M, N, K);
}

// ---------------- LSTM cell (one block per row m of the 16 bidir rows) ----------------
// pre   : [16][4096] = h_{t-1} @ Ucat^T          (gate order f,i,o,c at 0,1H,2H,3H)
// gate_t: [16][4096] = x-projections + bias for timestep t
// LayerNorm on the c-gate pre-activation uses unbiased std (ddof=1), eps added to std.
__global__ void k_cell(const float* __restrict__ pre, const float* __restrict__ gate_t,
                       const float* __restrict__ gamma, const float* __restrict__ beta,
                       float* __restrict__ c, float* __restrict__ hs_t,
                       bf16_t* __restrict__ hb) {
    const int m = blockIdx.x;        // 0..15
    const int tid = threadIdx.x;     // 256 threads, 4 elements each
    __shared__ float sred[256];
    const float* pm = pre    + (long)m * kG4;
    const float* gm = gate_t + (long)m * kG4;

    float av[4]; float s = 0.0f;
#pragma unroll
    for (int j = 0; j < 4; ++j) {
        int n = tid + j * 256;
        av[j] = pm[3 * kH + n] + gm[3 * kH + n];
        s += av[j];
    }
    sred[tid] = s; __syncthreads();
    for (int st = 128; st > 0; st >>= 1) { if (tid < st) sred[tid] += sred[tid + st]; __syncthreads(); }
    const float mean = sred[0] * (1.0f / (float)kH);
    __syncthreads();

    float ss = 0.0f;
#pragma unroll
    for (int j = 0; j < 4; ++j) { float d = av[j] - mean; ss += d * d; }
    sred[tid] = ss; __syncthreads();
    for (int st = 128; st > 0; st >>= 1) { if (tid < st) sred[tid] += sred[tid + st]; __syncthreads(); }
    const float stdv = sqrtf(sred[0] * (1.0f / (float)(kH - 1)));
    const float inv = 1.0f / (stdv + kEps);
    __syncthreads();

#pragma unroll
    for (int j = 0; j < 4; ++j) {
        int n = tid + j * 256;
        float at = gamma[n] * (av[j] - mean) * inv + beta[n];
        float fg = sigmoidf_(pm[0 * kH + n] + gm[0 * kH + n]);
        float ig = sigmoidf_(pm[1 * kH + n] + gm[1 * kH + n]);
        float og = sigmoidf_(pm[2 * kH + n] + gm[2 * kH + n]);
        long ci = (long)m * kH + n;
        float cn = ig * tanhf(at) * kDrop + fg * c[ci];
        c[ci] = cn;
        float h = og * tanhf(cn);
        hs_t[ci] = h;
        hb[ci] = __float2bfloat16(h);
    }
}

// ---------------- log-softmax + loss/err (one block per (t,b) row) ----------------
// out[0]=loss (mean NLL), out[1]=err rate, out[2..]=pout [T][B][C]
__global__ void k_softmax_loss(const float* __restrict__ logits, const int* __restrict__ lab,
                               float* __restrict__ out) {
    const int row = blockIdx.x;      // 0..3199  == t*B + b
    const int tid = threadIdx.x;     // 256
    __shared__ float sval[256];
    __shared__ int   sidx[256];
    const float* lr = logits + (long)row * kC;

    float mx = -3.4e38f; int mi = 0;
    for (int n = tid; n < kC; n += 256) { float v = lr[n]; if (v > mx) { mx = v; mi = n; } }
    sval[tid] = mx; sidx[tid] = mi; __syncthreads();
    for (int st = 128; st > 0; st >>= 1) {
        if (tid < st) {
            if (sval[tid + st] > sval[tid] ||
                (sval[tid + st] == sval[tid] && sidx[tid + st] < sidx[tid])) {
                sval[tid] = sval[tid + st]; sidx[tid] = sidx[tid + st];
            }
        }
        __syncthreads();
    }
    const float rmax = sval[0]; const int amax = sidx[0];
    __syncthreads();

    float se = 0.0f;
    for (int n = tid; n < kC; n += 256) se += __expf(lr[n] - rmax);
    sval[tid] = se; __syncthreads();
    for (int st = 128; st > 0; st >>= 1) { if (tid < st) sval[tid] += sval[tid + st]; __syncthreads(); }
    const float lse = rmax + logf(sval[0]);

    float* prow = out + 2 + (long)row * kC;
    for (int n = tid; n < kC; n += 256) prow[n] = lr[n] - lse;

    if (tid == 0) {
        const int l = lab[row];
        atomicAdd(&out[0], -(lr[l] - lse) * (1.0f / (float)kRows));
        atomicAdd(&out[1], (amax != l) ? (1.0f / (float)kRows) : 0.0f);
    }
}

// ---------------- host orchestration ----------------
extern "C" void kernel_launch(void* const* d_in, const int* in_sizes, int n_in,
                              void* d_out, int out_size, void* d_ws, size_t ws_size,
                              hipStream_t stream) {
    (void)in_sizes; (void)n_in; (void)out_size; (void)ws_size;

    // Input order: x, lab, test_flag, then params pytree in jax sorted-key order:
    // top: 'bo', 'layers'(list of 3 dicts), 'wo'
    // per-layer sorted keys: bc, beta, bf, bi, bo, gamma, uc, uf, ui, uo, wc, wf, wi, wo
    const float* x   = (const float*)d_in[0];
    const int*   lab = (const int*)d_in[1];
    const float* bo_top = (const float*)d_in[3];
    struct Layer { const float *bc,*beta,*bf,*bi,*bo,*gamma,*uc,*uf,*ui,*uo,*wc,*wf,*wi,*wo; };
    Layer L[3];
    int idx = 4;
    for (int l = 0; l < 3; ++l) {
        L[l].bc   = (const float*)d_in[idx++]; L[l].beta = (const float*)d_in[idx++];
        L[l].bf   = (const float*)d_in[idx++]; L[l].bi   = (const float*)d_in[idx++];
        L[l].bo   = (const float*)d_in[idx++]; L[l].gamma= (const float*)d_in[idx++];
        L[l].uc   = (const float*)d_in[idx++]; L[l].uf   = (const float*)d_in[idx++];
        L[l].ui   = (const float*)d_in[idx++]; L[l].uo   = (const float*)d_in[idx++];
        L[l].wc   = (const float*)d_in[idx++]; L[l].wf   = (const float*)d_in[idx++];
        L[l].wi   = (const float*)d_in[idx++]; L[l].wo   = (const float*)d_in[idx++];
    }
    const float* wo_top = (const float*)d_in[idx];

    // workspace carve-up (256B aligned)
    char* ws = (char*)d_ws; size_t off = 0;
    auto carve = [&](size_t bytes) -> void* {
        void* p = ws + off; off += (bytes + 255) & ~(size_t)255; return p;
    };
    bf16_t* xb     = (bf16_t*)carve((size_t)kT * kB2 * kD1 * 2);       // bidir bf16 input (max ld)
    float*  gates  = (float*) carve((size_t)kT * kB2 * kG4 * 4);       // x-projections, all t
    float*  pre    = (float*) carve((size_t)kB2 * kG4 * 4);            // h@U^T for one step
    float*  cbuf   = (float*) carve((size_t)kB2 * kH * 4);             // cell state
    bf16_t* hb     = (bf16_t*)carve((size_t)kB2 * kH * 2);             // h bf16 (GEMM input)
    float*  hs     = (float*) carve((size_t)kT * kB2 * kH * 4);        // all hidden states
    float*  xA     = (float*) carve((size_t)kT * kB * kD1 * 4);        // layer output ping
    float*  xBuf   = (float*) carve((size_t)kT * kB * kD1 * 4);        // layer output pong
    bf16_t* wb     = (bf16_t*)carve((size_t)4 * kH * kD1 * 2);         // Wcat bf16 [4096][Kpad]
    bf16_t* ub     = (bf16_t*)carve((size_t)kG4 * kH * 2);             // Ucat bf16 [4096][1024]
    bf16_t* wob    = (bf16_t*)carve((size_t)kC * kD1 * 2);             // classifier W bf16
    float*  bcat   = (float*) carve((size_t)kG4 * 4);                  // concat gate bias
    float*  logits = (float*) carve((size_t)kRows * kC * 4);

    const int CVT_B = 2048, THR = 256;

    const float* xin = x;
    int Din = kD0, Kpad = kD0P;
    float* xnext = xA;

    for (int l = 0; l < 3; ++l) {
        // bidirectional bf16 (K-padded) input for this layer
        k_build_xb<<<CVT_B, THR, 0, stream>>>(xin, xb, Din, Kpad);

        // convert weights (gate order f,i,o,c), concatenated along rows
        const float* Wsrc[4] = { L[l].wf, L[l].wi, L[l].wo, L[l].wc };
        const float* Usrc[4] = { L[l].uf, L[l].ui, L[l].uo, L[l].uc };
        for (int g = 0; g < 4; ++g) {
            k_w_pad_bf16<<<CVT_B, THR, 0, stream>>>(Wsrc[g], wb + (size_t)g * kH * Kpad, kH, Din, Kpad);
            k_w_pad_bf16<<<CVT_B, THR, 0, stream>>>(Usrc[g], ub + (size_t)g * kH * kH, kH, kH, kH);
        }
        k_bias_concat<<<kG4 / THR, THR, 0, stream>>>(L[l].bf, L[l].bi, L[l].bo, L[l].bc, bcat);

        // gate input projections for every timestep, all 4 gates in one wide GEMM:
        // [6400, Kpad] x [Kpad, 4096]  (Wcat rows are f,i,o,c blocks)
        launch_gemm<4>(xb, Kpad, wb, Kpad, bcat, gates, kG4, kT * kB2, kG4, Kpad, stream);

        // recurrence (latency-critical: keep 1 tile/wave for max parallelism; U is L2-resident)
        hipMemsetAsync(cbuf, 0, (size_t)kB2 * kH * 4, stream);
        hipMemsetAsync(hb,   0, (size_t)kB2 * kH * 2, stream);
        for (int t = 0; t < kT; ++t) {
            launch_gemm<1>(hb, kH, ub, kH, nullptr, pre, kG4, kB2, kG4, kH, stream);
            k_cell<<<kB2, 256, 0, stream>>>(pre, gates + (size_t)t * kB2 * kG4,
                                            L[l].gamma, L[l].beta,
                                            cbuf, hs + (size_t)t * kB2 * kH, hb);
        }

        // merge fwd/bwd halves into next-layer input [T][B][2H]
        k_merge_bidir<<<CVT_B, THR, 0, stream>>>(hs, xnext);
        xin = xnext; xnext = (xnext == xA) ? xBuf : xA;
        Din = kD1; Kpad = kD1;
    }

    // classifier: logits = x_final @ wo^T + bo
    k_w_pad_bf16<<<CVT_B, THR, 0, stream>>>(wo_top, wob, kC, kD1, kD1);
    k_f32_to_bf16<<<CVT_B, THR, 0, stream>>>(xin, xb, (long)kRows * kD1);
    launch_gemm<4>(xb, kD1, wob, kD1, bo_top, logits, kC, kRows, kC, kD1, stream);

    // loss / err / pout
    hipMemsetAsync(d_out, 0, 2 * sizeof(float), stream);
    k_softmax_loss<<<kRows, 256, 0, stream>>>(logits, lab, (float*)d_out);
}
